// Cross_Att_54374285967791
// MI455X (gfx1250) — matrix operand order repaired and checked by reading, not compiled, not run on hardware
//
#include <hip/hip_runtime.h>
#include <cstdint>
#include <cstddef>

// ---------------------------------------------------------------------------
// Types for CDNA5 WMMA (wave32): v_wmma_f32_16x16x32_bf16
// ---------------------------------------------------------------------------
typedef __attribute__((ext_vector_type(16))) __bf16 v16bf;
typedef __attribute__((ext_vector_type(8)))  float  v8f;
typedef __attribute__((ext_vector_type(4)))  unsigned uint4v;

union BFrag { uint4v q[2]; v16bf v; };

#define BATCH 16
#define TOK   1024
#define TOK1  1025
#define MROWS (BATCH*TOK1)   // 16400 real token rows
#define MPAD  16416          // 513 * 32 : GEMM M padded to the 32-row wave tile

__device__ __forceinline__ unsigned short f2bf(float x) {
  unsigned u = __float_as_uint(x);
  unsigned r = u + 0x7FFFu + ((u >> 16) & 1u);   // round-to-nearest-even
  return (unsigned short)(r >> 16);
}

// ---------------------------------------------------------------------------
// Generic GEMM:  Out[M,N] = A[M,K](bf16) x W[N,K](bf16)^T, fp32 accum.
// Wave tile: 32(M) x 64(N) -> 8 accumulators; A frags reused 4x, B frags 2x.
// K-loop software-pipelined: next k-step fragments prefetched before the
// current 8 WMMAs so loads have a full iteration of latency slack.
// Epilogue MODE: 0 = f32 store
//               1 = sigmoid -> f32
//               2 = relu^2  -> bf16
//               3 = res + acc -> f32          (residual add, in-place safe)
//               4 = res + gate*acc -> f32     (gate is pre-sigmoided)
// ---------------------------------------------------------------------------
template<int MODE>
__global__ __launch_bounds__(128) void gemm_wmma(
    const unsigned short* __restrict__ A,
    const unsigned short* __restrict__ W,
    float*          __restrict__ outF,
    unsigned short* __restrict__ outB,
    const float*    __restrict__ res,
    const float*    __restrict__ gate,
    int M, int N, int K)
{
  const int lane   = threadIdx.x;
  const int mTiles = M >> 5;
  const int nTiles = N >> 6;
  const int id = blockIdx.x * 4 + threadIdx.y;
  if (id >= mTiles * nTiles) return;
  const int mT  = id % mTiles;
  const int nT  = id / mTiles;
  const int l15 = lane & 15;
  const int hi  = lane >> 4;
  const int aOff = hi ? 8 : 0;    // A frag: lanes 0-15 -> K 0..7,16..23 ; 16-31 -> 8..15,24..31
  const int bOff = hi ? 8 : 0;    // B frag: lanes 0-15 -> K 0..7,16..23 ; 16-31 -> 8..15,24..31

  const unsigned short* A0 = A + (size_t)((mT << 5) + l15) * K + aOff;
  const unsigned short* A1 = A0 + (size_t)16 * K;
  const unsigned short* Wp = W + (size_t)((nT << 6) + l15) * K + bOff;
  const size_t WS = (size_t)16 * K;

  v8f acc[8] = {};
  BFrag a0c, a1c, a0n, a1n, bc[4], bn[4];

  // prologue: k = 0 fragments
  a0c.q[0] = *(const uint4v*)(A0);       a0c.q[1] = *(const uint4v*)(A0 + 16);
  a1c.q[0] = *(const uint4v*)(A1);       a1c.q[1] = *(const uint4v*)(A1 + 16);
#pragma unroll
  for (int s = 0; s < 4; ++s) {
    const unsigned short* p = Wp + WS * s;
    bc[s].q[0] = *(const uint4v*)(p);    bc[s].q[1] = *(const uint4v*)(p + 16);
  }

  for (int k = 32; k < K; k += 32) {
    // prefetch next k-step
    a0n.q[0] = *(const uint4v*)(A0 + k); a0n.q[1] = *(const uint4v*)(A0 + k + 16);
    a1n.q[0] = *(const uint4v*)(A1 + k); a1n.q[1] = *(const uint4v*)(A1 + k + 16);
#pragma unroll
    for (int s = 0; s < 4; ++s) {
      const unsigned short* p = Wp + WS * s + k;
      bn[s].q[0] = *(const uint4v*)(p);  bn[s].q[1] = *(const uint4v*)(p + 16);
    }
    // compute on current fragments
#pragma unroll
    for (int s = 0; s < 4; ++s) {
      acc[s]     = __builtin_amdgcn_wmma_f32_16x16x32_bf16(
                       false, a0c.v, false, bc[s].v, (short)0, acc[s],     false, false);
      acc[4 + s] = __builtin_amdgcn_wmma_f32_16x16x32_bf16(
                       false, a1c.v, false, bc[s].v, (short)0, acc[4 + s], false, false);
    }
    a0c = a0n; a1c = a1n;
#pragma unroll
    for (int s = 0; s < 4; ++s) bc[s] = bn[s];
  }
  // tail k-step
#pragma unroll
  for (int s = 0; s < 4; ++s) {
    acc[s]     = __builtin_amdgcn_wmma_f32_16x16x32_bf16(
                     false, a0c.v, false, bc[s].v, (short)0, acc[s],     false, false);
    acc[4 + s] = __builtin_amdgcn_wmma_f32_16x16x32_bf16(
                     false, a1c.v, false, bc[s].v, (short)0, acc[4 + s], false, false);
  }

  // epilogue: C/D layout -> VGPR i = row i (+8 for hi lane half)
  const int mBase = (mT << 5) + (hi ? 8 : 0);
#pragma unroll
  for (int half = 0; half < 2; ++half) {
#pragma unroll
    for (int s = 0; s < 4; ++s) {
      const int nn = (nT << 6) + (s << 4) + l15;
#pragma unroll
      for (int i = 0; i < 8; ++i) {
        size_t o = (size_t)(mBase + half * 16 + i) * N + nn;
        float v = acc[half * 4 + s][i];
        if (MODE == 0)      outF[o] = v;
        else if (MODE == 1) outF[o] = 1.0f / (1.0f + __expf(-v));
        else if (MODE == 2) { float t = fmaxf(v, 0.0f); outB[o] = f2bf(t * t); }
        else if (MODE == 3) outF[o] = res[o] + v;
        else                outF[o] = res[o] + gate[o] * v;
      }
    }
  }
}

// ---------------------------------------------------------------------------
// Row LayerNorm over [rows, C].  Optional f32 and/or bf16 outputs.
// ---------------------------------------------------------------------------
template<int WF32, int WBF>
__global__ __launch_bounds__(256) void ln_rows(
    const float* __restrict__ in, const float* __restrict__ g,
    const float* __restrict__ be, float* __restrict__ outF,
    unsigned short* __restrict__ outB, int C)
{
  const int row = blockIdx.x;
  const float* x = in + (size_t)row * C;
  __shared__ float red[256];
  float s = 0.f, s2 = 0.f;
  for (int c = threadIdx.x; c < C; c += 256) { float v = x[c]; s += v; s2 += v * v; }
  red[threadIdx.x] = s; __syncthreads();
  for (int o = 128; o > 0; o >>= 1) { if (threadIdx.x < o) red[threadIdx.x] += red[threadIdx.x + o]; __syncthreads(); }
  const float mean = red[0] / C; __syncthreads();
  red[threadIdx.x] = s2; __syncthreads();
  for (int o = 128; o > 0; o >>= 1) { if (threadIdx.x < o) red[threadIdx.x] += red[threadIdx.x + o]; __syncthreads(); }
  const float var = red[0] / C - mean * mean;
  const float rs = rsqrtf(var + 1e-5f);
  for (int c = threadIdx.x; c < C; c += 256) {
    float v = (x[c] - mean) * rs * g[c] + be[c];
    if (WF32) outF[(size_t)row * C + c] = v;
    if (WBF)  outB[(size_t)row * C + c] = f2bf(v);
  }
}

// LN + mean-pool over tokens (cross stage). Reads X rows [b*T1 + 1 + t].
__global__ __launch_bounds__(256) void ln_pool(
    const float* __restrict__ X, const float* __restrict__ g,
    const float* __restrict__ be, float* __restrict__ pool, int C)
{
  const int b = blockIdx.x / TOK;
  const int t = blockIdx.x % TOK;
  const float* x = X + ((size_t)b * TOK1 + 1 + t) * C;
  __shared__ float red[256];
  float s = 0.f, s2 = 0.f;
  for (int c = threadIdx.x; c < C; c += 256) { float v = x[c]; s += v; s2 += v * v; }
  red[threadIdx.x] = s; __syncthreads();
  for (int o = 128; o > 0; o >>= 1) { if (threadIdx.x < o) red[threadIdx.x] += red[threadIdx.x + o]; __syncthreads(); }
  const float mean = red[0] / C; __syncthreads();
  red[threadIdx.x] = s2; __syncthreads();
  for (int o = 128; o > 0; o >>= 1) { if (threadIdx.x < o) red[threadIdx.x] += red[threadIdx.x + o]; __syncthreads(); }
  const float var = red[0] / C - mean * mean;
  const float rs = rsqrtf(var + 1e-5f);
  const float invT = 1.0f / TOK;
  for (int c = threadIdx.x; c < C; c += 256) {
    float v = (x[c] - mean) * rs * g[c] + be[c];
    atomicAdd(&pool[(size_t)b * C + c], v * invT);
  }
}

// ---------------------------------------------------------------------------
// Numerically-stable causal WKV scan, fused with sigmoid(r) gate, bf16 out.
// One thread per (batch, channel); sequential over T1 tokens.
// ---------------------------------------------------------------------------
__global__ __launch_bounds__(256) void wkv_scan(
    const float* __restrict__ kbuf, const float* __restrict__ vbuf,
    const float* __restrict__ srbuf, const float* __restrict__ decay,
    const float* __restrict__ first, unsigned short* __restrict__ ybf, int C)
{
  const int idx = blockIdx.x * 256 + threadIdx.x;
  if (idx >= BATCH * C) return;
  const int b = idx / C;
  const int c = idx % C;
  const float w = -__expf(decay[c] / (float)TOK1);
  const float u = first[c] / (float)TOK1;
  float a = 0.f, bb = 0.f, p = -1e38f;
  const size_t base = (size_t)b * TOK1 * C + c;
  for (int t = 0; t < TOK1; ++t) {
    const size_t o = base + (size_t)t * C;
    const float kt = kbuf[o], vt = vbuf[o];
    float ww = u + kt;
    float q = fmaxf(p, ww);
    float e1 = __expf(p - q), e2 = __expf(ww - q);
    float y = (e1 * a + e2 * vt) / (e1 * bb + e2);
    float ww2 = p + w;
    float q2 = fmaxf(ww2, kt);
    float f1 = __expf(ww2 - q2), f2 = __expf(kt - q2);
    a = f1 * a + f2 * vt; bb = f1 * bb + f2; p = q2;
    ybf[o] = f2bf(srbuf[o] * y);
  }
}

// ---------------------------------------------------------------------------
// Utility kernels
// ---------------------------------------------------------------------------
__global__ void f32_to_bf16_k(const float* __restrict__ in, unsigned short* __restrict__ out, size_t n) {
  size_t i = (size_t)blockIdx.x * 256 + threadIdx.x;
  if (i < n) out[i] = f2bf(in[i]);
}

__global__ void zero_f32(float* __restrict__ p, int n) {
  int i = blockIdx.x * 256 + threadIdx.x;
  if (i < n) p[i] = 0.f;
}

// src [B,C,T] -> X rows 1..T per batch  ([B,T1,C], row 0 left for cross token)
__global__ void chw_to_tokens(const float* __restrict__ src, float* __restrict__ X, int C) {
  size_t i = (size_t)blockIdx.x * 256 + threadIdx.x;
  size_t tot = (size_t)BATCH * C * TOK;
  if (i >= tot) return;
  int c = (int)(i % C);
  size_t rest = i / C;
  int t = (int)(rest % TOK);
  int b = (int)(rest / TOK);
  X[((size_t)b * TOK1 + 1 + t) * C + c] = src[((size_t)b * C + c) * TOK + t];
}

// X rows 1..T -> out [B,C,T]
__global__ void tokens_to_chw(const float* __restrict__ X, float* __restrict__ out, int C) {
  size_t i = (size_t)blockIdx.x * 256 + threadIdx.x;
  size_t tot = (size_t)BATCH * C * TOK;
  if (i >= tot) return;
  int t = (int)(i % TOK);
  size_t rest = i / TOK;
  int c = (int)(rest % C);
  int b = (int)(rest / C);
  out[i] = X[((size_t)b * TOK1 + 1 + t) * C + c];
}

// Cross projection: X[b, token0, d] = sum_c pool[b,c] * W[d,c] + bias[d]
__global__ void gemv_token0(const float* __restrict__ pool, const float* __restrict__ W,
                            const float* __restrict__ bias, float* __restrict__ X,
                            int Cin, int Dout) {
  int idx = blockIdx.x * 128 + threadIdx.x;
  if (idx >= BATCH * Dout) return;
  int b = idx / Dout;
  int d = idx % Dout;
  float s = bias[d];
  const float* pr = pool + (size_t)b * Cin;
  const float* wr = W + (size_t)d * Cin;
  for (int c = 0; c < Cin; ++c) s += pr[c] * wr[c];
  X[((size_t)b * TOK1) * Dout + d] = s;
}

// ---------------------------------------------------------------------------
// Host-side plumbing
// ---------------------------------------------------------------------------
enum { LN0G, LN0B, LN1G, LN1B, LN2G, LN2B, DECAY, FIRST,
       ATTK, ATTV, ATTR, ATTO, FFNK, FFNR, FFNV };
enum { NSG, NSB, NLG, NLB, LSW, LSB, LLW, LLB };

struct Bufs {
  float *F1, *F2, *F3, *F4;
  unsigned short *H0, *H1, *H2, *WB;
};

template<int MODE>
static void launch_gemm(const unsigned short* A, const unsigned short* W,
                        float* oF, unsigned short* oB, const float* res,
                        const float* gate, int M, int N, int K, hipStream_t s) {
  int tiles = (M >> 5) * (N >> 6);
  dim3 blk(32, 4);
  gemm_wmma<MODE><<<(tiles + 3) / 4, blk, 0, s>>>(A, W, oF, oB, res, gate, M, N, K);
}

static void run_block(const float* const* p, float* X, const Bufs& bu, int C, hipStream_t s) {
  const int MP = MPAD;      // padded GEMM M (pad rows compute garbage, never read)
  const int M  = MROWS;     // real rows for LN / WKV
  const int H  = 4 * C;
  const size_t C2 = (size_t)C * C;
  unsigned short* WB = bu.WB;
  // Convert weights to bf16 (L2-resident thereafter: all weights < 20 MB)
  auto cv = [&](const float* src, unsigned short* dst, size_t n) {
    f32_to_bf16_k<<<(unsigned)((n + 255) / 256), 256, 0, s>>>(src, dst, n);
  };
  cv(p[ATTK], WB,            C2);
  cv(p[ATTV], WB + C2,       C2);
  cv(p[ATTR], WB + 2 * C2,   C2);
  cv(p[ATTO], WB + 3 * C2,   C2);
  cv(p[FFNK], WB + 4 * C2, 4*C2);
  cv(p[FFNR], WB + 8 * C2,   C2);
  cv(p[FFNV], WB + 9 * C2, 4*C2);

  // ln0: X -> F1 (f32)
  ln_rows<1, 0><<<M, 256, 0, s>>>(X, p[LN0G], p[LN0B], bu.F1, nullptr, C);
  // ln1: F1 -> H0 (bf16, GEMM A)
  ln_rows<0, 1><<<M, 256, 0, s>>>(bu.F1, p[LN1G], p[LN1B], nullptr, bu.H0, C);
  // k, v, sigmoid(r)
  launch_gemm<0>(bu.H0, WB,          bu.F2, nullptr, nullptr, nullptr, MP, C, C, s);
  launch_gemm<0>(bu.H0, WB + C2,     bu.F3, nullptr, nullptr, nullptr, MP, C, C, s);
  launch_gemm<1>(bu.H0, WB + 2 * C2, bu.F4, nullptr, nullptr, nullptr, MP, C, C, s);
  // wkv fused with gate -> H1 (bf16)
  wkv_scan<<<(BATCH * C + 255) / 256, 256, 0, s>>>(bu.F2, bu.F3, bu.F4, p[DECAY], p[FIRST], bu.H1, C);
  // att_o + residual (in-place on F1)
  launch_gemm<3>(bu.H1, WB + 3 * C2, bu.F1, nullptr, bu.F1, nullptr, MP, C, C, s);
  // ln2: F1 -> H0 (bf16)
  ln_rows<0, 1><<<M, 256, 0, s>>>(bu.F1, p[LN2G], p[LN2B], nullptr, bu.H0, C);
  // ffn_k -> relu^2 -> H2 (bf16) ; ffn_r -> sigmoid -> F2
  launch_gemm<2>(bu.H0, WB + 4 * C2, nullptr, bu.H2, nullptr, nullptr, MP, H, C, s);
  launch_gemm<1>(bu.H0, WB + 8 * C2, bu.F2, nullptr, nullptr, nullptr, MP, C, C, s);
  // ffn_v:  X = F1 + F2 * (H2 x Wv^T)
  launch_gemm<4>(bu.H2, WB + 9 * C2, X, nullptr, bu.F1, bu.F2, MP, C, H, s);
}

extern "C" void kernel_launch(void* const* d_in, const int* in_sizes, int n_in,
                              void* d_out, int out_size, void* d_ws, size_t ws_size,
                              hipStream_t stream) {
  (void)in_sizes; (void)n_in; (void)out_size; (void)ws_size;
  const int CS = 384, CL = 768;
  const float* e = (const float*)d_in[0];
  const float* r = (const float*)d_in[1];
  const float* const* ps = (const float* const*)(d_in + 2);    // params_s (15)
  const float* const* pl = (const float* const*)(d_in + 17);   // params_l (15)
  const float* const* cr = (const float* const*)(d_in + 32);   // cross (8)

  // ---- workspace carve-out (all M-dim buffers sized for MPAD rows) ----
  char* base = (char*)d_ws;
  size_t off = 0;
  auto alloc = [&](size_t bytes) -> void* {
    void* p = base + off;
    off += (bytes + 255) & ~(size_t)255;
    return p;
  };
  float* XS = (float*)alloc((size_t)MPAD * CS * 4);
  float* XL = (float*)alloc((size_t)MPAD * CL * 4);
  Bufs bu;
  bu.F1 = (float*)alloc((size_t)MPAD * CL * 4);
  bu.F2 = (float*)alloc((size_t)MPAD * CL * 4);
  bu.F3 = (float*)alloc((size_t)MPAD * CL * 4);
  bu.F4 = (float*)alloc((size_t)MPAD * CL * 4);
  bu.H0 = (unsigned short*)alloc((size_t)MPAD * CL * 2);
  bu.H1 = (unsigned short*)alloc((size_t)MPAD * CL * 2);
  bu.H2 = (unsigned short*)alloc((size_t)MPAD * 4 * CL * 2);
  bu.WB = (unsigned short*)alloc((size_t)13 * CL * CL * 2);
  float* poolE = (float*)alloc((size_t)BATCH * CL * 4);
  float* poolR = (float*)alloc((size_t)BATCH * CS * 4);

  const size_t totL = (size_t)BATCH * CL * TOK;
  const size_t totS = (size_t)BATCH * CS * TOK;

  // ---- cross stage ----
  chw_to_tokens<<<(unsigned)((totS + 255) / 256), 256, 0, stream>>>(r, XS, CS);
  chw_to_tokens<<<(unsigned)((totL + 255) / 256), 256, 0, stream>>>(e, XL, CL);
  zero_f32<<<(BATCH * CL + 255) / 256, 256, 0, stream>>>(poolE, BATCH * CL);
  zero_f32<<<(BATCH * CS + 255) / 256, 256, 0, stream>>>(poolR, BATCH * CS);
  ln_pool<<<BATCH * TOK, 256, 0, stream>>>(XL, cr[NLG], cr[NLB], poolE, CL);
  ln_pool<<<BATCH * TOK, 256, 0, stream>>>(XS, cr[NSG], cr[NSB], poolR, CS);
  // e_t @ lin_l_w.T + lin_l_b -> token0 of XS ;  r_t @ lin_s_w.T + lin_s_b -> token0 of XL
  gemv_token0<<<(BATCH * CS + 127) / 128, 128, 0, stream>>>(poolE, cr[LLW], cr[LLB], XS, CL, CS);
  gemv_token0<<<(BATCH * CL + 127) / 128, 128, 0, stream>>>(poolR, cr[LSW], cr[LSB], XL, CS, CL);

  // ---- small block (writes r_out), then large block (writes e_out) ----
  float* e_out = (float*)d_out;
  float* r_out = e_out + (size_t)BATCH * CL * TOK;

  run_block(ps, XS, bu, CS, stream);
  tokens_to_chw<<<(unsigned)((totS + 255) / 256), 256, 0, stream>>>(XS, r_out, CS);

  run_block(pl, XL, bu, CL, stream);
  tokens_to_chw<<<(unsigned)((totL + 255) / 256), 256, 0, stream>>>(XL, e_out, CL);
}
